// BrainGAT_45243185496499
// MI455X (gfx1250) — matrix operand order repaired and checked by reading, not compiled, not run on hardware
//
#include <hip/hip_runtime.h>
#include <hip/hip_bf16.h>
#include <math.h>

#define NN     16384
#define EE     65536
#define ETOT   (EE + NN)
#define HH     8
#define CC     256
#define DD     2048       // H*C
#define BB     64
#define INCH   116
#define EPSBN  1e-5f

typedef __attribute__((ext_vector_type(16))) __bf16 v16bf;
typedef __attribute__((ext_vector_type(8)))  float  v8f;

// ---------------- helpers ----------------
__device__ __forceinline__ unsigned short f2bf(float x) {
    unsigned int u = __float_as_uint(x);
    unsigned int r = (u + 0x7FFFu + ((u >> 16) & 1u)) >> 16;   // RNE
    return (unsigned short)r;
}

__device__ __forceinline__ float atomicMaxF(float* addr, float val) {
    int* ai = (int*)addr;
    int old = __float_as_int(*addr);
    while (__int_as_float(old) < val) {
        int assumed = old;
        old = atomicCAS(ai, assumed, __float_as_int(val));
        if (old == assumed) break;
    }
    return __int_as_float(old);
}

// A fragment from permuted LDS layout: lane's 16 bf16 stored contiguously
// -> two ds_load_b128 per fragment.
__device__ __forceinline__ v16bf load_fragA_perm(const unsigned short* lds, int tile, int lane) {
    const unsigned int* p = (const unsigned int*)lds + (tile * 32 + lane) * 8;
    union { v16bf v; uint4 q[2]; } f;
    f.q[0] = *(const uint4*)p;
    f.q[1] = *(const uint4*)(p + 4);
    return f.v;
}

// B fragment: LDS holds W tile transposed as [col][k] (32 bf16 per col).
// lanes 0-15: K=0..7,16..23, lanes 16-31: K=8..15,24..31 -> b128 pairs.
__device__ __forceinline__ v16bf load_fragB(const unsigned short* lds, int colBase, int lane) {
    const unsigned int* p = (const unsigned int*)lds +
                            (colBase + (lane & 15)) * 16 + ((lane >> 4) ? 4 : 0);
    union { v16bf v; uint4 q[2]; } f;
    f.q[0] = *(const uint4*)p;
    f.q[1] = *(const uint4*)(p + 8);
    return f.v;
}

// ---------------- WMMA GEMM: C[M,Nout] = A[M,K] * W[K,Nout]
// 128x128 block tile, 256 threads = 8 waves (2 row-groups x 4 col-groups),
// each wave: 4 M-tiles x 2 N-strips = 8 x v_wmma_f32_16x16x32_bf16 per K-step.
// BOUNDS=false requires K % 32 == 0 (vectorized b128 loads, b64 LDS stores).
template<bool BOUNDS>
__global__ void __launch_bounds__(256)
gat_gemm_bf16(const float* __restrict__ A, const float* __restrict__ W,
              float* __restrict__ Cmat, int M, int K, int Nout) {
    __shared__ __align__(16) unsigned short lA[128 * 32];  // permuted fragment order
    __shared__ __align__(16) unsigned short lB[128 * 32];  // [col][k]
    const int tid = threadIdx.x;
    const int wv = tid >> 5, ln = tid & 31;
    const int wr = wv >> 2;            // 0..1 : 64-row group
    const int wc = wv & 3;             // 0..3 : 32-col group
    const int bc = blockIdx.x * 128, br = blockIdx.y * 128;

    v8f acc[4][2];
#pragma unroll
    for (int i = 0; i < 4; ++i)
#pragma unroll
        for (int j = 0; j < 2; ++j)
#pragma unroll
            for (int r = 0; r < 8; ++r) acc[i][j][r] = 0.0f;

    for (int kk = 0; kk < K; kk += 32) {
        if (BOUNDS) {
            // scalar, bounds-checked path (layer 1, K=116)
#pragma unroll
            for (int i = 0; i < 16; ++i) {
                int lin = i * 256 + tid;                 // 0..4095
                int r = lin >> 5, c = lin & 31;
                float v = (kk + c < K) ? A[(size_t)(br + r) * K + kk + c] : 0.0f;
                int tile = r >> 4, m = r & 15;
                int q = c >> 1, lo = c & 1;
                int half = (q >> 2) & 1;
                int pos = (q & 3) + ((q >> 3) << 2);
                lA[((tile * 32 + half * 16 + m) * 8 + pos) * 2 + lo] = f2bf(v);
            }
#pragma unroll
            for (int i = 0; i < 16; ++i) {
                int lin = i * 256 + tid;
                int k = lin >> 7, n = lin & 127;
                float v = (kk + k < K) ? W[(size_t)(kk + k) * Nout + bc + n] : 0.0f;
                lB[n * 32 + k] = f2bf(v);
            }
        } else {
            // vectorized path: A via float4 (global_load_b128), LDS via b64 stores
#pragma unroll
            for (int i = 0; i < 4; ++i) {
                int lin = i * 256 + tid;                 // quad index 0..1023
                int r = lin >> 3, c = (lin & 7) * 4;     // 4 consecutive K
                float4 v = *(const float4*)(A + (size_t)(br + r) * K + kk + c);
                unsigned int u0 = (unsigned int)f2bf(v.x) | ((unsigned int)f2bf(v.y) << 16);
                unsigned int u1 = (unsigned int)f2bf(v.z) | ((unsigned int)f2bf(v.w) << 16);
                int tile = r >> 4, m = r & 15;
                int q = c >> 1;                          // even; q and q+1 same half
                int half = (q >> 2) & 1;
                int pos = (q & 3) + ((q >> 3) << 2);     // even
                unsigned int* dst = (unsigned int*)lA + (tile * 32 + half * 16 + m) * 8 + pos;
                uint2 st; st.x = u0; st.y = u1;
                *(uint2*)dst = st;                       // ds_store_b64
            }
            // B: thread covers (col n, K-quad); lanes sweep n -> coalesced
#pragma unroll
            for (int i = 0; i < 4; ++i) {
                int lin = i * 256 + tid;                 // 0..1023 quads
                int n = lin & 127, k0 = (lin >> 7) * 4;
                const float* wp = W + (size_t)(kk + k0) * Nout + bc + n;
                float a0 = wp[0];
                float a1 = wp[Nout];
                float a2 = wp[2 * (size_t)Nout];
                float a3 = wp[3 * (size_t)Nout];
                unsigned int u0 = (unsigned int)f2bf(a0) | ((unsigned int)f2bf(a1) << 16);
                unsigned int u1 = (unsigned int)f2bf(a2) | ((unsigned int)f2bf(a3) << 16);
                unsigned int* dst = (unsigned int*)lB + n * 16 + k0 / 2;  // k0/2 even
                uint2 st; st.x = u0; st.y = u1;
                *(uint2*)dst = st;                       // ds_store_b64
            }
        }
        __syncthreads();

        v16bf bf0 = load_fragB(lB, wc * 32, ln);
        v16bf bf1 = load_fragB(lB, wc * 32 + 16, ln);
#pragma unroll
        for (int mt = 0; mt < 4; ++mt) {
            v16bf af = load_fragA_perm(lA, wr * 4 + mt, ln);
            acc[mt][0] = __builtin_amdgcn_wmma_f32_16x16x32_bf16(
                false, af, false, bf0, (short)0, acc[mt][0], false, false);
            acc[mt][1] = __builtin_amdgcn_wmma_f32_16x16x32_bf16(
                false, af, false, bf1, (short)0, acc[mt][1], false, false);
        }
        __syncthreads();
    }

    // C/D layout: VGPR r -> lanes 0-15: M=r, lanes 16-31: M=r+8; N=lane&15
    const int rofs = (ln >> 4) ? 8 : 0;
#pragma unroll
    for (int mt = 0; mt < 4; ++mt) {
        int rowb = br + wr * 64 + mt * 16 + rofs;
#pragma unroll
        for (int s = 0; s < 2; ++s) {
            int col = bc + wc * 32 + s * 16 + (ln & 15);
#pragma unroll
            for (int r = 0; r < 8; ++r)
                Cmat[(size_t)(rowb + r) * Nout + col] = acc[mt][s][r];
        }
    }
}

// ---------------- small kernels ----------------
__global__ void k_reduce_mean(const float* __restrict__ x, int n, float* out) {
    __shared__ float s[256];
    float a = 0.f;
    for (int i = threadIdx.x; i < n; i += 256) a += x[i];
    s[threadIdx.x] = a; __syncthreads();
    for (int o = 128; o > 0; o >>= 1) { if (threadIdx.x < o) s[threadIdx.x] += s[threadIdx.x + o]; __syncthreads(); }
    if (threadIdx.x == 0) out[0] = s[0] / (float)n;
}

__global__ void k_wedot(const float* __restrict__ We, const float* __restrict__ ae, float* wd) {
    __shared__ float s[256];
    int h = blockIdx.x, t = threadIdx.x;
    s[t] = We[h * CC + t] * ae[h * CC + t]; __syncthreads();
    for (int o = 128; o > 0; o >>= 1) { if (t < o) s[t] += s[t + o]; __syncthreads(); }
    if (t == 0) wd[h] = s[0];
}

__global__ void __launch_bounds__(256)
k_scores(const float* __restrict__ xh, const float* __restrict__ asrc,
         const float* __restrict__ adst, float* __restrict__ ss, float* __restrict__ sd) {
    int n = blockIdx.x, t = threadIdx.x;
    int h = t >> 5, ln = t & 31;
    const float* xr = xh + (size_t)n * DD + h * CC;
    float s1 = 0.f, s2 = 0.f;
    for (int c = ln; c < CC; c += 32) {
        float v = xr[c];
        s1 += v * asrc[h * CC + c];
        s2 += v * adst[h * CC + c];
    }
    for (int o = 16; o > 0; o >>= 1) { s1 += __shfl_xor(s1, o, 32); s2 += __shfl_xor(s2, o, 32); }
    if (ln == 0) { ss[n * HH + h] = s1; sd[n * HH + h] = s2; }
}

__global__ void k_fill(float* p, int n, float v) {
    int i = blockIdx.x * 256 + threadIdx.x;
    if (i < n) p[i] = v;
}

__global__ void k_edge_max(const int* __restrict__ ei, const float* __restrict__ ea,
                           const float* __restrict__ emean, const float* __restrict__ wd,
                           const float* __restrict__ ss, const float* __restrict__ sd,
                           float* __restrict__ att, float* __restrict__ m) {
    int i = blockIdx.x * 256 + threadIdx.x;
    if (i >= ETOT * HH) return;
    int e = i >> 3, h = i & 7;
    int s, dn; float attr;
    if (e < EE) { s = ei[e]; dn = ei[EE + e]; attr = ea[e]; }
    else        { s = dn = e - EE; attr = emean[0]; }
    float a = ss[s * HH + h] + sd[dn * HH + h] + attr * wd[h];
    a = (a > 0.f) ? a : 0.2f * a;               // leaky_relu(0.2)
    att[i] = a;
    atomicMaxF(&m[dn * HH + h], a);
}

__global__ void k_edge_sum(const int* __restrict__ ei, float* __restrict__ att,
                           const float* __restrict__ m, float* __restrict__ den) {
    int i = blockIdx.x * 256 + threadIdx.x;
    if (i >= ETOT * HH) return;
    int e = i >> 3, h = i & 7;
    int dn = (e < EE) ? ei[EE + e] : (e - EE);
    float ex = __expf(att[i] - m[dn * HH + h]);
    att[i] = ex;
    atomicAdd(&den[dn * HH + h], ex);
}

__global__ void __launch_bounds__(256)
k_agg(const int* __restrict__ ei, const float* __restrict__ att,
      const float* __restrict__ den, const float* __restrict__ xh, float* __restrict__ out) {
    __shared__ float alpha[HH];
    __shared__ int sh_s, sh_d;
    int e = blockIdx.x;
    if (threadIdx.x == 0) {
        if (e < EE) { sh_s = ei[e]; sh_d = ei[EE + e]; }
        else        { sh_s = sh_d = e - EE; }
    }
    __syncthreads();
    if (threadIdx.x < HH)
        alpha[threadIdx.x] = att[(size_t)e * HH + threadIdx.x] /
                             (den[sh_d * HH + threadIdx.x] + 1e-16f);
    __syncthreads();
    const float* xs = xh + (size_t)sh_s * DD;
    float* od = out + (size_t)sh_d * DD;
#pragma unroll
    for (int j = 0; j < 8; ++j) {
        int c = j * 256 + threadIdx.x;                 // head j, channel threadIdx.x
        atomicAdd(&od[c], alpha[j] * xs[c]);
    }
}

__global__ void k_add_bias(float* __restrict__ x, const float* __restrict__ bias, int D, int tot) {
    int i = blockIdx.x * 256 + threadIdx.x;
    if (i < tot) x[i] += bias[i % D];
}

__global__ void k_headmean(const float* __restrict__ in, const float* __restrict__ bias,
                           float* __restrict__ out) {
    int i = blockIdx.x * 256 + threadIdx.x;           // i over N*CC
    if (i >= NN * CC) return;
    int n = i >> 8, c = i & 255;
    const float* row = in + (size_t)n * DD;
    float a = 0.f;
#pragma unroll
    for (int h = 0; h < HH; ++h) a += row[h * CC + c];
    out[i] = a * (1.0f / HH) + bias[c];
}

__global__ void k_colsum(const float* __restrict__ x, float* __restrict__ s,
                         float* __restrict__ sq, int D) {
    int col = blockIdx.x * 256 + threadIdx.x;
    int r0 = blockIdx.y * 256;
    float a = 0.f, b = 0.f;
    for (int r = r0; r < r0 + 256; ++r) {
        float v = x[(size_t)r * D + col];
        a += v; b += v * v;
    }
    atomicAdd(&s[col], a); atomicAdd(&sq[col], b);
}

__global__ void k_bn_elu(float* __restrict__ x, const float* __restrict__ s,
                         const float* __restrict__ sq, const float* __restrict__ g,
                         const float* __restrict__ b, int D, int tot) {
    int i = blockIdx.x * 256 + threadIdx.x;
    if (i >= tot) return;
    int col = i % D;
    float mu = s[col] * (1.0f / NN);
    float var = sq[col] * (1.0f / NN) - mu * mu;
    var = fmaxf(var, 0.f);
    float v = (x[i] - mu) * rsqrtf(var + EPSBN) * g[col] + b[col];
    x[i] = (v > 0.f) ? v : (__expf(v) - 1.0f);        // ELU
}

__global__ void k_pool(const float* __restrict__ h, const int* __restrict__ batch,
                       float* __restrict__ psum, float* __restrict__ pmax,
                       float* __restrict__ cnt) {
    int n = blockIdx.x, c = threadIdx.x;
    int b = batch[n];
    float v = h[(size_t)n * CC + c];
    atomicAdd(&psum[b * CC + c], v);
    atomicMaxF(&pmax[b * CC + c], v);
    if (c == 0) atomicAdd(&cnt[b], 1.0f);
}

__global__ void k_poolfin(const float* __restrict__ psum, const float* __restrict__ pmax,
                          const float* __restrict__ cnt, float* __restrict__ gfeat) {
    int i = blockIdx.x * 256 + threadIdx.x;           // over B*CC
    if (i >= BB * CC) return;
    int b = i >> 8, c = i & 255;
    float cc = cnt[b];
    gfeat[b * (2 * CC) + c] = psum[i] / fmaxf(cc, 1.0f);
    gfeat[b * (2 * CC) + CC + c] = (cc > 0.f) ? pmax[i] : 0.0f;
}

__global__ void k_fc(const float* __restrict__ x, const float* __restrict__ W,
                     const float* __restrict__ bias, float* __restrict__ y,
                     int Din, int Dout) {
    int i = blockIdx.x * 256 + threadIdx.x;
    if (i >= BB * Dout) return;
    int r = i / Dout, c = i % Dout;
    float a = bias[c];
    for (int k = 0; k < Din; ++k) a += x[r * Din + k] * W[k * Dout + c];
    y[i] = a;
}

__global__ void k_bn_relu64(const float* __restrict__ x, const float* __restrict__ g,
                            const float* __restrict__ b, float* __restrict__ y, int D) {
    __shared__ float s1[64], s2[64];
    int col = blockIdx.x, t = threadIdx.x;
    float v = x[(size_t)t * D + col];
    s1[t] = v; s2[t] = v * v; __syncthreads();
    for (int o = 32; o > 0; o >>= 1) {
        if (t < o) { s1[t] += s1[t + o]; s2[t] += s2[t + o]; }
        __syncthreads();
    }
    float mu = s1[0] * (1.0f / BB);
    float var = fmaxf(s2[0] * (1.0f / BB) - mu * mu, 0.f);
    float r = (v - mu) * rsqrtf(var + EPSBN) * g[col] + b[col];
    y[(size_t)t * D + col] = fmaxf(r, 0.f);
}

__global__ void k_fc3(const float* __restrict__ x, const float* __restrict__ W,
                      const float* __restrict__ bias, float* __restrict__ y) {
    int i = threadIdx.x;                               // 128 = 64 rows x 2 classes
    if (i >= BB * 2) return;
    int r = i >> 1, c = i & 1;
    float a = bias[c];
    for (int k = 0; k < 128; ++k) a += x[r * 128 + k] * W[k * 2 + c];
    y[i] = a;
}

// ---------------- host launcher ----------------
extern "C" void kernel_launch(void* const* d_in, const int* in_sizes, int n_in,
                              void* d_out, int out_size, void* d_ws, size_t ws_size,
                              hipStream_t stream) {
    (void)in_sizes; (void)n_in; (void)out_size; (void)ws_size;

    const float* x_in  = (const float*)d_in[0];
    const int*   ei    = (const int*)d_in[1];
    const float* ea    = (const float*)d_in[2];
    const int*   batch = (const int*)d_in[3];
    const float* c1_W = (const float*)d_in[28]; const float* c1_b = (const float*)d_in[29];
    const float* c1_g = (const float*)d_in[30]; const float* c1_bb = (const float*)d_in[31];
    const float* c2_W = (const float*)d_in[32]; const float* c2_b = (const float*)d_in[33];
    const float* c2_g = (const float*)d_in[34]; const float* c2_bb = (const float*)d_in[35];
    const float* c3_W = (const float*)d_in[36]; const float* c3_b = (const float*)d_in[37];

    // workspace layout
    char* w = (char*)d_ws;
    float* bufA   = (float*)w; w += (size_t)NN * DD * 4;     // 128 MB
    float* bufB   = (float*)w; w += (size_t)NN * DD * 4;     // 128 MB
    float* att    = (float*)w; w += (size_t)ETOT * HH * 4;   // 2.6 MB
    float* ssrc   = (float*)w; w += (size_t)NN * HH * 4;
    float* sdst   = (float*)w; w += (size_t)NN * HH * 4;
    float* mbuf   = (float*)w; w += (size_t)NN * HH * 4;
    float* den    = (float*)w; w += (size_t)NN * HH * 4;
    float* colsum = (float*)w; w += DD * 4;
    float* colsq  = (float*)w; w += DD * 4;
    float* wedot  = (float*)w; w += 256;
    float* emean  = (float*)w; w += 256;
    float* psum   = (float*)w; w += (size_t)BB * CC * 4;
    float* pmax   = (float*)w; w += (size_t)BB * CC * 4;
    float* cnt    = (float*)w; w += 256;
    float* gfeat  = (float*)w; w += (size_t)BB * 2 * CC * 4;
    float* h1pre  = (float*)w; w += (size_t)BB * CC * 4;
    float* h1     = (float*)w; w += (size_t)BB * CC * 4;
    float* h2pre  = (float*)w; w += (size_t)BB * 128 * 4;
    float* h2     = (float*)w; w += (size_t)BB * 128 * 4;

    const int eblk = (ETOT * HH + 255) / 256;
    const float NEGINF = -INFINITY;

    k_reduce_mean<<<1, 256, 0, stream>>>(ea, EE, emean);

    for (int L = 0; L < 3; ++L) {
        const int base = 4 + 8 * L;
        const float* W_   = (const float*)d_in[base + 0];
        const float* aS   = (const float*)d_in[base + 1];
        const float* aD   = (const float*)d_in[base + 2];
        const float* We_  = (const float*)d_in[base + 3];
        const float* aE   = (const float*)d_in[base + 4];
        const float* bias = (const float*)d_in[base + 5];
        const float* bnG  = (const float*)d_in[base + 6];
        const float* bnB  = (const float*)d_in[base + 7];

        // xh = Ain @ W   (bufB)
        if (L == 0)
            gat_gemm_bf16<true><<<dim3(DD / 128, NN / 128), 256, 0, stream>>>(
                x_in, W_, bufB, NN, INCH, DD);
        else
            gat_gemm_bf16<false><<<dim3(DD / 128, NN / 128), 256, 0, stream>>>(
                bufA, W_, bufB, NN, DD, DD);

        k_wedot<<<HH, 256, 0, stream>>>(We_, aE, wedot);
        k_scores<<<NN, 256, 0, stream>>>(bufB, aS, aD, ssrc, sdst);

        hipMemsetAsync(den, 0, (size_t)NN * HH * 4, stream);
        hipMemsetAsync(bufA, 0, (size_t)NN * DD * 4, stream);
        k_fill<<<(NN * HH + 255) / 256, 256, 0, stream>>>(mbuf, NN * HH, NEGINF);

        k_edge_max<<<eblk, 256, 0, stream>>>(ei, ea, emean, wedot, ssrc, sdst, att, mbuf);
        k_edge_sum<<<eblk, 256, 0, stream>>>(ei, att, mbuf, den);
        k_agg<<<ETOT, 256, 0, stream>>>(ei, att, den, bufB, bufA);

        if (L < 2) {
            k_add_bias<<<(NN * DD + 255) / 256, 256, 0, stream>>>(bufA, bias, DD, NN * DD);
            hipMemsetAsync(colsum, 0, DD * 4, stream);
            hipMemsetAsync(colsq, 0, DD * 4, stream);
            k_colsum<<<dim3(DD / 256, NN / 256), 256, 0, stream>>>(bufA, colsum, colsq, DD);
            k_bn_elu<<<(NN * DD + 255) / 256, 256, 0, stream>>>(bufA, colsum, colsq, bnG, bnB, DD, NN * DD);
        } else {
            // head mean + bias -> bufB[N,256], then BN+ELU
            k_headmean<<<(NN * CC + 255) / 256, 256, 0, stream>>>(bufA, bias, bufB);
            hipMemsetAsync(colsum, 0, CC * 4, stream);
            hipMemsetAsync(colsq, 0, CC * 4, stream);
            k_colsum<<<dim3(CC / 256, NN / 256), 256, 0, stream>>>(bufB, colsum, colsq, CC);
            k_bn_elu<<<(NN * CC + 255) / 256, 256, 0, stream>>>(bufB, colsum, colsq, bnG, bnB, CC, NN * CC);
        }
    }

    // global pooling (h3 in bufB[N,256])
    hipMemsetAsync(psum, 0, (size_t)BB * CC * 4, stream);
    hipMemsetAsync(cnt, 0, BB * 4, stream);
    k_fill<<<(BB * CC + 255) / 256, 256, 0, stream>>>(pmax, BB * CC, NEGINF);
    k_pool<<<NN, CC, 0, stream>>>(bufB, batch, psum, pmax, cnt);
    k_poolfin<<<(BB * CC + 255) / 256, 256, 0, stream>>>(psum, pmax, cnt, gfeat);

    // classifier
    k_fc<<<(BB * CC + 255) / 256, 256, 0, stream>>>(gfeat, c1_W, c1_b, h1pre, 2 * CC, CC);
    k_bn_relu64<<<CC, 64, 0, stream>>>(h1pre, c1_g, c1_bb, h1, CC);
    k_fc<<<(BB * 128 + 255) / 256, 256, 0, stream>>>(h1, c2_W, c2_b, h2pre, CC, 128);
    k_bn_relu64<<<128, 64, 0, stream>>>(h2pre, c2_g, c2_bb, h2, 128);
    k_fc3<<<1, 128, 0, stream>>>(h2, c3_W, c3_b, (float*)d_out);
}
